// GQASelfAttention_47433618817469
// MI455X (gfx1250) — matrix operand order repaired and checked by reading, not compiled, not run on hardware
//
#include <hip/hip_runtime.h>

typedef __attribute__((ext_vector_type(16))) _Float16 v16h;
typedef __attribute__((ext_vector_type(8)))  _Float16 v8h;
typedef __attribute__((ext_vector_type(8)))  float    v8f;
typedef __attribute__((ext_vector_type(4)))  unsigned int u32x4;
typedef __attribute__((ext_vector_type(8)))  unsigned int u32x8;

union V16 { v16h v; v8h h[2]; };

#define B_SZ   4
#define S_LEN  2048
#define E_DIM  1024
#define KV_DIM 256
#define HEADS  16
#define GROUPS 4

#define USE_TDM 1   // attention K/V tile staging via Tensor Data Mover

// packed fp32->f16 convert (v_cvt_pk_rtz_f16_f32)
#define CVTPK(a, b) __builtin_amdgcn_cvt_pkrtz((a), (b))
typedef decltype(__builtin_amdgcn_cvt_pkrtz(0.f, 0.f)) pk2;

// ---------------------------------------------------------------------------
// Tiled f16 WMMA GEMM:  C[M,N] = (A[M,K] @ W[K,N] + bias) * scale
// Block tile 128x128, 256 threads = 8 waves, wave tile 32x64, K-step 32.
// AF16: A is _Float16 (else fp32, converted while staging).
// OMODE 0: store f16 row-major [M,N]
// OMODE 1: store f16 KV-transposed: n=(g,d), m=(b,s) -> out[((b*G+g)*64+d)*S + s]
// OMODE 2: store fp32 row-major [M,N]
// ---------------------------------------------------------------------------
template<bool AF16, int OMODE>
__global__ __launch_bounds__(256) void gemm_wmma(
    const void* __restrict__ Aptr, const float* __restrict__ W,
    const float* __restrict__ bias, void* __restrict__ Cptr,
    int M, int N, int K, float scale)
{
  __shared__ __align__(16) _Float16 Ah[128 * 32];   // [m][k]
  __shared__ __align__(16) _Float16 Wt[128 * 32];   // [n][k]  (transposed W tile)

  const int tid  = threadIdx.x;
  const int wave = tid >> 5;
  const int lane = tid & 31;
  const int hi   = lane >> 4;      // which 16-lane half
  const int ln   = lane & 15;
  const int bm   = blockIdx.y * 128;
  const int bn   = blockIdx.x * 128;
  const int wm   = (wave >> 1) * 32;   // wave row offset in block tile
  const int wn   = (wave & 1) * 64;    // wave col offset in block tile

  v8f acc[2][4] = {};                  // 2x4 subtiles of 16x16, f32 accum

  const int sm = tid >> 1, skc = (tid & 1) * 16;      // A staging: row, k-chunk
  const int wkp = (tid & 15) * 2, wng = (tid >> 4) * 8; // W staging: k-pair, n-group

  for (int k0 = 0; k0 < K; k0 += 32) {
    __syncthreads();
    // ---- stage A tile (128 x 32) into LDS as f16 ----
    {
      _Float16* dst = Ah + sm * 32 + skc;
      if constexpr (AF16) {
        const _Float16* src = (const _Float16*)Aptr + (size_t)(bm + sm) * K + k0 + skc;
        *(v8h*)dst       = *(const v8h*)src;
        *(v8h*)(dst + 8) = *(const v8h*)(src + 8);
        if (k0 + 32 < K) __builtin_prefetch(src + 32, 0, 1);
      } else {
        const float* src = (const float*)Aptr + (size_t)(bm + sm) * K + k0 + skc;
#pragma unroll
        for (int i = 0; i < 16; i += 8) {
          float4 f0 = *(const float4*)(src + i);
          float4 f1 = *(const float4*)(src + i + 4);
          pk2 a0 = CVTPK(f0.x, f0.y), a1 = CVTPK(f0.z, f0.w);
          pk2 a2 = CVTPK(f1.x, f1.y), a3 = CVTPK(f1.z, f1.w);
          auto q0 = __builtin_shufflevector(a0, a1, 0, 1, 2, 3);
          auto q1 = __builtin_shufflevector(a2, a3, 0, 1, 2, 3);
          auto o8 = __builtin_shufflevector(q0, q1, 0, 1, 2, 3, 4, 5, 6, 7);
          *(decltype(o8)*)(dst + i) = o8;
        }
        if (k0 + 32 < K) __builtin_prefetch(src + 32, 0, 1);
      }
    }
    // ---- stage W tile transposed (32 x 128 -> Wt[n][k]), packed k-pairs ----
    {
      const float* r0 = W + (size_t)(k0 + wkp) * N + bn + wng;
      const float* r1 = r0 + N;
      float4 f0a = *(const float4*)(r0);
      float4 f0b = *(const float4*)(r0 + 4);
      float4 f1a = *(const float4*)(r1);
      float4 f1b = *(const float4*)(r1 + 4);
      const float fa[8] = { f0a.x, f0a.y, f0a.z, f0a.w, f0b.x, f0b.y, f0b.z, f0b.w };
      const float fb[8] = { f1a.x, f1a.y, f1a.z, f1a.w, f1b.x, f1b.y, f1b.z, f1b.w };
#pragma unroll
      for (int i = 0; i < 8; ++i) {
        pk2 p = CVTPK(fa[i], fb[i]);                    // (k, k+1) pair for column n
        *(pk2*)(Wt + (wng + i) * 32 + wkp) = p;
      }
      if (k0 + 32 < K) __builtin_prefetch(r0 + 32 * N, 0, 1);
    }
    __syncthreads();

    // ---- build fragments (A and B both use contiguous k = hi*16 + 0..15) ----
    v16h af[2], bf[4];
#pragma unroll
    for (int i = 0; i < 2; ++i) {
      const _Float16* ab = Ah + (wm + i * 16 + ln) * 32;
      V16 u;
      u.h[0] = *(const v8h*)(ab + hi * 16);       // K = hi*16 + 0..7
      u.h[1] = *(const v8h*)(ab + hi * 16 + 8);   // K = hi*16 + 8..15
      af[i] = u.v;
    }
#pragma unroll
    for (int j = 0; j < 4; ++j) {
      const _Float16* bb = Wt + (wn + j * 16 + ln) * 32 + hi * 16;  // K = hi*16 + 0..15
      V16 u;
      u.h[0] = *(const v8h*)bb;
      u.h[1] = *(const v8h*)(bb + 8);
      bf[j] = u.v;
    }
    // ---- 8 WMMAs per wave per K-step ----
#pragma unroll
    for (int i = 0; i < 2; ++i)
#pragma unroll
      for (int j = 0; j < 4; ++j)
        acc[i][j] = __builtin_amdgcn_wmma_f32_16x16x32_f16(
            false, af[i], false, bf[j], (short)0, acc[i][j], false, false);
  }

  // ---- epilogue: bias, scale, store (C/D layout: VGPRr -> M=r+hi*8, lane%16 -> N) ----
#pragma unroll
  for (int i = 0; i < 2; ++i) {
#pragma unroll
    for (int j = 0; j < 4; ++j) {
      const int nn = bn + wn + j * 16 + ln;
      const float bv = bias[nn];
#pragma unroll
      for (int r = 0; r < 8; ++r) {
        const int mm = bm + wm + i * 16 + r + hi * 8;
        float val = (acc[i][j][r] + bv) * scale;
        if constexpr (OMODE == 0) {
          ((_Float16*)Cptr)[(size_t)mm * N + nn] = (_Float16)val;
        } else if constexpr (OMODE == 1) {
          const int b = mm >> 11, s = mm & 2047;      // S = 2048
          const int g = nn >> 6,  d = nn & 63;        // D = 64
          ((_Float16*)Cptr)[((size_t)(b * GROUPS + g) * 64 + d) * S_LEN + s] = (_Float16)val;
        } else {
          ((float*)Cptr)[(size_t)mm * N + nn] = val;
        }
      }
    }
  }
}

#if USE_TDM
// Issue a TDM 2D tile load: rows x cols (f16), row stride in elements.
// Must be executed by one wave; completion via s_wait_tensorcnt.
__device__ __forceinline__ void tdm_load_tile_f16(
    unsigned lds_byte_addr, const _Float16* gptr,
    unsigned tile_cols, unsigned tile_rows, unsigned row_stride_elems)
{
  const unsigned long long ga = (unsigned long long)(size_t)gptr;
  u32x4 g0;
  g0[0] = 1u;                                   // count=1, user descriptor
  g0[1] = lds_byte_addr;                        // lds_addr (bytes)
  g0[2] = (unsigned)ga;                         // global_addr[95:64]
  g0[3] = ((unsigned)(ga >> 32) & 0x01FFFFFFu)  // global_addr[120:96]
        | (2u << 30);                           // type = 2 ("image")
  u32x8 g1;
  g1[0] = 1u << 16;                             // workgroup_mask=0, data_size=1 (2B)
  g1[1] = (0x100000u & 0xFFFFu) << 16;          // tensor_dim0 lo16 (1M elems, no OOB)
  g1[2] = (0x100000u >> 16)                     // tensor_dim0 hi16
        | ((0x100000u & 0xFFFFu) << 16);        // tensor_dim1 lo16
  g1[3] = (0x100000u >> 16)                     // tensor_dim1 hi16
        | (tile_cols << 16);                    // tile_dim0
  g1[4] = tile_rows;                            // tile_dim1 (tile_dim2 = 0)
  g1[5] = row_stride_elems;                     // tensor_dim0_stride lo32
  g1[6] = 0u;                                   // stride0 hi16, stride1 lo16
  g1[7] = 0u;                                   // stride1 hi32
  asm volatile("tensor_load_to_lds %0, %1" :: "s"(g0), "s"(g1) : "memory");
}
#endif

// ---------------------------------------------------------------------------
// Flash attention: grid (S/64, B*HEADS), 128 threads = 4 waves x 16 q-rows.
// q: f16 [B,S,E] (pre-scaled by 1/sqrt(D)), k: f16 [B,S,KV_DIM],
// vt: f16 [B,G,D,S] (pre-transposed), o: f16 [B,S,E]
// ---------------------------------------------------------------------------
__global__ __launch_bounds__(128) void attn_wmma(
    const _Float16* __restrict__ q, const _Float16* __restrict__ k,
    const _Float16* __restrict__ vt, _Float16* __restrict__ o)
{
  __shared__ __align__(16) _Float16 Kt[64 * 64];       // [key][d]
  __shared__ __align__(16) _Float16 Vt[64 * 64];       // [d][key]
  __shared__ __align__(16) _Float16 Pb[4][16 * 64];    // per-wave P scratch [q][key]

  const int tid  = threadIdx.x;
  const int wave = tid >> 5;
  const int lane = tid & 31;
  const int hi   = lane >> 4;
  const int ln   = lane & 15;

  const int bh = blockIdx.y;        // b*16 + h
  const int b  = bh >> 4;
  const int h  = bh & 15;
  const int g  = h >> 2;            // KV group (h = g*4 + r)
  const int q0 = blockIdx.x * 64 + wave * 16;

  // ---- load Q fragments once (16 rows x 64 d = two 16x32 A-slabs) ----
  const _Float16* qrow = q + ((size_t)(b * S_LEN) + q0 + ln) * E_DIM + h * 64;
  v16h qa[2];
#pragma unroll
  for (int c = 0; c < 2; ++c) {
    V16 u;
    u.h[0] = *(const v8h*)(qrow + c * 32 + hi * 16);
    u.h[1] = *(const v8h*)(qrow + c * 32 + hi * 16 + 8);
    qa[c] = u.v;
  }

  v8f   oacc[4] = {};
  float mrow[8], lrow[8];
#pragma unroll
  for (int r = 0; r < 8; ++r) { mrow[r] = -1e30f; lrow[r] = 0.0f; }

#if !USE_TDM
  const int srow = tid >> 1, sch = (tid & 1) * 32;   // staging row / 32-half chunk
#endif

  for (int kt = 0; kt < S_LEN / 64; ++kt) {
    const int key0 = kt * 64;
    __syncthreads();
    // ---- stage K tile (64 keys x 64 d) and Vt tile (64 d x 64 keys) ----
#if USE_TDM
    if (wave == 0) {
      const _Float16* ksrc = k + ((size_t)(b * S_LEN) + key0) * KV_DIM + g * 64;
      const _Float16* vsrc = vt + ((size_t)(b * GROUPS + g) * 64) * S_LEN + key0;
      tdm_load_tile_f16((unsigned)(size_t)Kt, ksrc, 64u, 64u, KV_DIM);
      tdm_load_tile_f16((unsigned)(size_t)Vt, vsrc, 64u, 64u, S_LEN);
      __builtin_amdgcn_s_wait_tensorcnt(0);
    }
#else
    {
      const _Float16* ks = k + ((size_t)(b * S_LEN) + key0 + srow) * KV_DIM + g * 64 + sch;
      v8h* kd = (v8h*)(Kt + srow * 64 + sch);
      const v8h* ks8 = (const v8h*)ks;
      kd[0] = ks8[0]; kd[1] = ks8[1]; kd[2] = ks8[2]; kd[3] = ks8[3];

      const _Float16* vs = vt + ((size_t)(b * GROUPS + g) * 64 + srow) * S_LEN + key0 + sch;
      v8h* vd = (v8h*)(Vt + srow * 64 + sch);
      const v8h* vs8 = (const v8h*)vs;
      vd[0] = vs8[0]; vd[1] = vs8[1]; vd[2] = vs8[2]; vd[3] = vs8[3];
    }
#endif
    __syncthreads();

    // ---- scores: 4 key-subtiles x 2 k-steps = 8 WMMAs ----
    float sc[4][8];
#pragma unroll
    for (int s4 = 0; s4 < 4; ++s4) {
      v8f a = {};
#pragma unroll
      for (int c = 0; c < 2; ++c) {
        const _Float16* bp = Kt + (s4 * 16 + ln) * 64 + c * 32 + hi * 16;
        V16 u; u.h[0] = *(const v8h*)bp; u.h[1] = *(const v8h*)(bp + 8);
        a = __builtin_amdgcn_wmma_f32_16x16x32_f16(false, qa[c], false, u.v,
                                                   (short)0, a, false, false);
      }
#pragma unroll
      for (int r = 0; r < 8; ++r) sc[s4][r] = a[r];
    }

    // ---- online softmax (row-wise reduce across 16-lane halves) ----
#pragma unroll
    for (int r = 0; r < 8; ++r) {
      float mx = fmaxf(fmaxf(sc[0][r], sc[1][r]), fmaxf(sc[2][r], sc[3][r]));
#pragma unroll
      for (int msk = 1; msk < 16; msk <<= 1)
        mx = fmaxf(mx, __shfl_xor(mx, msk, 32));
      const float mnew  = fmaxf(mrow[r], mx);
      const float alpha = __expf(mrow[r] - mnew);
      mrow[r] = mnew;
      float rs = 0.0f;
#pragma unroll
      for (int s4 = 0; s4 < 4; ++s4) {
        float p = __expf(sc[s4][r] - mnew);
        sc[s4][r] = p;
        rs += p;
      }
#pragma unroll
      for (int msk = 1; msk < 16; msk <<= 1)
        rs += __shfl_xor(rs, msk, 32);
      lrow[r] = lrow[r] * alpha + rs;
#pragma unroll
      for (int d4 = 0; d4 < 4; ++d4) oacc[d4][r] *= alpha;
    }

    // ---- D-layout -> A-layout via per-wave LDS round trip ----
    _Float16* pw = Pb[wave];
#pragma unroll
    for (int s4 = 0; s4 < 4; ++s4)
#pragma unroll
      for (int r = 0; r < 8; ++r)
        pw[(r + hi * 8) * 64 + s4 * 16 + ln] = (_Float16)sc[s4][r];

    v16h pa[2];
#pragma unroll
    for (int c = 0; c < 2; ++c) {
      const _Float16* pr = pw + ln * 64 + c * 32 + hi * 16;
      V16 u; u.h[0] = *(const v8h*)pr; u.h[1] = *(const v8h*)(pr + 8);
      pa[c] = u.v;
    }

    // ---- O += P @ V : 4 d-subtiles x 2 key-steps = 8 WMMAs ----
#pragma unroll
    for (int d4 = 0; d4 < 4; ++d4)
#pragma unroll
      for (int c = 0; c < 2; ++c) {
        const _Float16* vp = Vt + (d4 * 16 + ln) * 64 + c * 32 + hi * 16;
        V16 u; u.h[0] = *(const v8h*)vp; u.h[1] = *(const v8h*)(vp + 8);
        oacc[d4] = __builtin_amdgcn_wmma_f32_16x16x32_f16(false, pa[c], false, u.v,
                                                          (short)0, oacc[d4], false, false);
      }
  }

  // ---- normalize and store f16 output [B,S,E] ----
#pragma unroll
  for (int d4 = 0; d4 < 4; ++d4)
#pragma unroll
    for (int r = 0; r < 8; ++r) {
      const float val = oacc[d4][r] / lrow[r];
      o[((size_t)(b * S_LEN) + q0 + r + hi * 8) * E_DIM + h * 64 + d4 * 16 + ln] =
          (_Float16)val;
    }
}

// ---------------------------------------------------------------------------
extern "C" void kernel_launch(void* const* d_in, const int* in_sizes, int n_in,
                              void* d_out, int out_size, void* d_ws, size_t ws_size,
                              hipStream_t stream) {
  const float* x  = (const float*)d_in[0];
  const float* Wq = (const float*)d_in[1];
  const float* bq = (const float*)d_in[2];
  const float* Wk = (const float*)d_in[3];
  const float* bk = (const float*)d_in[4];
  const float* Wv = (const float*)d_in[5];
  const float* bv = (const float*)d_in[6];
  const float* Wo = (const float*)d_in[7];
  const float* bo = (const float*)d_in[8];
  float* out = (float*)d_out;

  const int M = B_SZ * S_LEN;   // 8192
  char* w = (char*)d_ws;
  _Float16* qh  = (_Float16*)w;                                             // 16 MB
  _Float16* kh  = (_Float16*)(w + (size_t)M * E_DIM * 2);                   //  4 MB
  _Float16* vth = (_Float16*)(w + (size_t)M * E_DIM * 2 +
                              (size_t)M * KV_DIM * 2);                      //  4 MB
  _Float16* ah  = (_Float16*)(w + (size_t)M * E_DIM * 2 +
                              2 * (size_t)M * KV_DIM * 2);                  // 16 MB

  const dim3 blk(256);
  // Q = (x @ Wq + bq) * 1/sqrt(64)  -> f16 [M, E]
  gemm_wmma<false, 0><<<dim3(E_DIM / 128, M / 128), blk, 0, stream>>>(
      x, Wq, bq, qh, M, E_DIM, E_DIM, 0.125f);
  // K = x @ Wk + bk -> f16 [M, KV]
  gemm_wmma<false, 0><<<dim3(KV_DIM / 128, M / 128), blk, 0, stream>>>(
      x, Wk, bk, kh, M, KV_DIM, E_DIM, 1.0f);
  // V^T = (x @ Wv + bv) transposed -> f16 [B, G, D, S]
  gemm_wmma<false, 1><<<dim3(KV_DIM / 128, M / 128), blk, 0, stream>>>(
      x, Wv, bv, vth, M, KV_DIM, E_DIM, 1.0f);
  // attention -> f16 [M, E]
  attn_wmma<<<dim3(S_LEN / 64, B_SZ * HEADS), dim3(128), 0, stream>>>(qh, kh, vth, ah);
  // out = attn @ Wo + bo -> fp32
  gemm_wmma<true, 2><<<dim3(E_DIM / 128, M / 128), blk, 0, stream>>>(
      ah, Wo, bo, out, M, E_DIM, E_DIM, 1.0f);
}
